// MultiHeadedAttention_31988916421161
// MI455X (gfx1250) — matrix operand order repaired and checked by reading, not compiled, not run on hardware
//
#include <hip/hip_runtime.h>

// ---------------- problem constants ----------------
#define N_EMB   1024
#define HEADS   16
#define HDIM    64
#define CTX     2048
#define BATCH   4
#define BT      (BATCH * CTX)          // 8192
#define SCALE   0.125f                 // 64^-0.5

// ---------------- WMMA types / helpers ----------------
typedef __attribute__((ext_vector_type(16))) __bf16        v16bf;
typedef __attribute__((ext_vector_type(8)))  float         v8f;
typedef __attribute__((ext_vector_type(8)))  unsigned int  v8u;

__device__ inline unsigned short f2bf_u16(float f) {
    unsigned u = __builtin_bit_cast(unsigned, f);
    unsigned r = u + 0x7FFFu + ((u >> 16) & 1u);   // round-to-nearest-even
    return (unsigned short)(r >> 16);
}

__device__ inline v8f zero8() {
    v8f z;
#pragma unroll
    for (int i = 0; i < 8; ++i) z[i] = 0.0f;
    return z;
}

// Async memory->LDS copy of 32 contiguous bytes (2 x B128), ASYNCcnt-tracked.
// INST_OFFSET is added to BOTH lds and global addresses (ISA 08 section 4.4),
// so the second op reuses the same address VGPRs with offset:16.
__device__ inline void async_copy_32B(const unsigned short* gsrc,
                                      unsigned short* ldst) {
    unsigned lds = (unsigned)(size_t)(void*)ldst;   // low 32 bits = LDS byte addr
    asm volatile("global_load_async_to_lds_b128 %0, %1, off"
                 :: "v"(lds), "v"(gsrc) : "memory");
    asm volatile("global_load_async_to_lds_b128 %0, %1, off offset:16"
                 :: "v"(lds), "v"(gsrc) : "memory");
}

__device__ inline void async_fence() {
    asm volatile("s_wait_asynccnt 0" ::: "memory");
}

// A-fragment, 16x32 bf16 (MxK), source row-major [row][K], ld in ushorts.
// lane: row = lane&15 ; K-elements: (lane>>4)*8 + {0..7}, then +16 for elems 8..15
__device__ inline v16bf load_a_frag(const unsigned short* p, int ld) {
    int lane = threadIdx.x & 31;
    int row  = lane & 15;
    int kh   = lane >> 4;
    const unsigned short* q = p + (size_t)row * ld + kh * 8;
    uint4 lo = *(const uint4*)q;          // K = kh*8 + 0..7
    uint4 hi = *(const uint4*)(q + 16);   // K = kh*8 + 16..23
    v8u u = {lo.x, lo.y, lo.z, lo.w, hi.x, hi.y, hi.z, hi.w};
    return __builtin_bit_cast(v16bf, u);
}

// B-fragment, 32x16 bf16 (KxN), source stored [N][K] (K contiguous), ld in ushorts.
// lane: n = lane&15 ; K-elements: (lane>>4)*8 + {0..7}, then +16 for elems 8..15
__device__ inline v16bf load_b_frag(const unsigned short* p, int ld) {
    int lane = threadIdx.x & 31;
    int n    = lane & 15;
    int kh   = lane >> 4;
    const unsigned short* q = p + (size_t)n * ld + kh * 8;
    uint4 lo = *(const uint4*)q;          // K = kh*8 + 0..7
    uint4 hi = *(const uint4*)(q + 16);   // K = kh*8 + 16..23
    v8u u = {lo.x, lo.y, lo.z, lo.w, hi.x, hi.y, hi.z, hi.w};
    return __builtin_bit_cast(v16bf, u);
}

__device__ inline v8f wmma_bf16(v16bf a, v16bf b, v8f c) {
    return __builtin_amdgcn_wmma_f32_16x16x32_bf16(false, a, false, b, (short)0, c,
                                                   false, false);
}

// ---------------- stage 0: conversions ----------------
__global__ void cvt_bf16_kernel(const float* __restrict__ in,
                                unsigned short* __restrict__ out, int n) {
    int i = blockIdx.x * blockDim.x + threadIdx.x;
    if (i < n) out[i] = f2bf_u16(in[i]);
}

// in [G][K][N] (f32) -> out [G][N][K] (bf16)
__global__ void cvt_transpose_kernel(const float* __restrict__ in,
                                     unsigned short* __restrict__ out,
                                     int G, int K, int N) {
    int i = blockIdx.x * blockDim.x + threadIdx.x;
    int total = G * K * N;
    if (i >= total) return;
    int g   = i / (K * N);
    int rem = i - g * (K * N);
    int kk  = rem / N;
    int n   = rem - kk * N;
    out[(size_t)g * N * K + (size_t)n * K + kk] = f2bf_u16(in[i]);
}

// ---------------- stage 1: QKV projection ----------------
// q,k: bf16 [b*H+h][t][d] ; vT: bf16 [b*H+h][d][t]
__global__ __launch_bounds__(256) void qkv_proj_kernel(
    const unsigned short* __restrict__ xb,
    const unsigned short* __restrict__ wqT,
    const unsigned short* __restrict__ wkT,
    const unsigned short* __restrict__ wvT,
    unsigned short* __restrict__ q,
    unsigned short* __restrict__ k,
    unsigned short* __restrict__ vT) {
    __shared__ __align__(16) unsigned short xs[128 * 48];
    __shared__ __align__(16) unsigned short ws[3 * 64 * 48];

    int tid  = threadIdx.x;
    int lane = tid & 31, wave = tid >> 5;
    int bh = blockIdx.y;
    int b = bh >> 4, h = bh & 15;
    int tbase = blockIdx.x * 128;

    v8f acc[3][4];
#pragma unroll
    for (int m = 0; m < 3; ++m)
#pragma unroll
        for (int nb = 0; nb < 4; ++nb) acc[m][nb] = zero8();

    for (int kc = 0; kc < N_EMB; kc += 32) {
        {   // x tile: 128 rows x 32 cols, 256 async tasks of 32B
            int row = tid >> 1, half = tid & 1;
            const unsigned short* src =
                xb + ((size_t)(b * CTX + tbase + row)) * N_EMB + kc + half * 16;
            async_copy_32B(src, &xs[row * 48 + half * 16]);
        }
        for (int task = tid; task < 384; task += 256) {  // 3 weight tiles 64x32
            int m = task >> 7;
            int rem = task & 127;
            int row = rem >> 1, half = rem & 1;
            const unsigned short* wp = (m == 0) ? wqT : ((m == 1) ? wkT : wvT);
            const unsigned short* src =
                wp + ((size_t)(h * HDIM + row)) * N_EMB + kc + half * 16;
            async_copy_32B(src, &ws[m * 64 * 48 + row * 48 + half * 16]);
        }
        async_fence();          // this wave's async->LDS writes have landed
        __syncthreads();

        v16bf a = load_a_frag(&xs[wave * 16 * 48], 48);
#pragma unroll
        for (int m = 0; m < 3; ++m)
#pragma unroll
            for (int nb = 0; nb < 4; ++nb) {
                v16bf bf = load_b_frag(&ws[m * 64 * 48 + nb * 16 * 48], 48);
                acc[m][nb] = wmma_bf16(a, bf, acc[m][nb]);
            }
        __syncthreads();
    }

    int lh = lane >> 4, col = lane & 15;
#pragma unroll
    for (int nb = 0; nb < 4; ++nb)
#pragma unroll
        for (int v2 = 0; v2 < 8; ++v2) {
            int t = tbase + wave * 16 + v2 + 8 * lh;
            int d = nb * 16 + col;
            size_t qi = ((size_t)bh * CTX + t) * HDIM + d;
            q[qi] = f2bf_u16(acc[0][nb][v2]);
            k[qi] = f2bf_u16(acc[1][nb][v2]);
            size_t vi = ((size_t)bh * HDIM + d) * CTX + t;
            vT[vi] = f2bf_u16(acc[2][nb][v2]);
        }
}

// ---------------- stage 2: column softmax stats ----------------
// softmax over t (query axis) per column s, causal t >= s.
__global__ __launch_bounds__(128) void col_stats_kernel(
    const unsigned short* __restrict__ q,
    const unsigned short* __restrict__ k,
    float* __restrict__ mbuf, float* __restrict__ rbuf) {
    int lane = threadIdx.x & 31, wave = threadIdx.x >> 5;
    int gw = blockIdx.x * 4 + wave;
    int sb = gw & 127;          // CTX/16 = 128 column blocks
    int bh = gw >> 7;
    int s0 = sb * 16;

    const unsigned short* kb = k + ((size_t)bh * CTX + s0) * HDIM;
    v16bf b0 = load_b_frag(kb, HDIM);        // d = 0..31
    v16bf b1 = load_b_frag(kb + 32, HDIM);   // d = 32..63

    int lh = lane >> 4, col = lane & 15;
    float m = -INFINITY, sum = 0.0f;

    for (int t0 = s0; t0 < CTX; t0 += 16) {
        const unsigned short* qb = q + ((size_t)bh * CTX + t0) * HDIM;
        if (t0 + 16 < CTX)      // prefetch next Q tile row block
            __builtin_prefetch(qb + 16 * HDIM, 0, 3);
        v16bf a0 = load_a_frag(qb, HDIM);
        v16bf a1 = load_a_frag(qb + 32, HDIM);
        v8f acc = zero8();
        acc = wmma_bf16(a0, b0, acc);
        acc = wmma_bf16(a1, b1, acc);

        float vals[8];
        float tmax = -INFINITY;
#pragma unroll
        for (int v2 = 0; v2 < 8; ++v2) {
            int t = t0 + v2 + 8 * lh;
            int s = s0 + col;
            float val = (t >= s) ? acc[v2] * SCALE : -INFINITY;
            vals[v2] = val;
            tmax = fmaxf(tmax, val);
        }
        tmax = fmaxf(tmax, __shfl_xor(tmax, 16, 32));
        float newm = fmaxf(m, tmax);
        float ts = 0.0f;
#pragma unroll
        for (int v2 = 0; v2 < 8; ++v2) ts += __expf(vals[v2] - newm);
        ts += __shfl_xor(ts, 16, 32);
        sum = sum * __expf(m - newm) + ts;
        m = newm;
    }
    if (lane < 16) {
        mbuf[(size_t)bh * CTX + s0 + lane] = m;
        rbuf[(size_t)bh * CTX + s0 + lane] = 1.0f / sum;
    }
}

// ---------------- stage 3: fused P = softmax(QK^T), O = P V ----------------
__global__ __launch_bounds__(128) void attn_out_kernel(
    const unsigned short* __restrict__ q,
    const unsigned short* __restrict__ k,
    const unsigned short* __restrict__ vT,
    const float* __restrict__ mbuf, const float* __restrict__ rbuf,
    unsigned short* __restrict__ attn) {
    __shared__ __align__(16) unsigned short pbuf[4][16 * 48];

    int lane = threadIdx.x & 31, wave = threadIdx.x >> 5;
    int gw = blockIdx.x * 4 + wave;
    int tb = gw & 127;
    int bh = gw >> 7;
    int t0 = tb * 16;
    int b = bh >> 4, h = bh & 15;

    const unsigned short* qb = q + ((size_t)bh * CTX + t0) * HDIM;
    v16bf a0 = load_a_frag(qb, HDIM);
    v16bf a1 = load_a_frag(qb + 32, HDIM);

    v8f acco[4];
#pragma unroll
    for (int nb = 0; nb < 4; ++nb) acco[nb] = zero8();

    int lh = lane >> 4, col = lane & 15;
    unsigned short* pb = pbuf[wave];

    for (int s0 = 0; s0 < t0 + 16; s0 += 32) {
        // prefetch next K tile while this one computes
        if (s0 + 32 < t0 + 16)
            __builtin_prefetch(k + ((size_t)bh * CTX + s0 + 32) * HDIM, 0, 3);
#pragma unroll
        for (int hh = 0; hh < 2; ++hh) {
            int sc = s0 + hh * 16;
            const unsigned short* kb = k + ((size_t)bh * CTX + sc) * HDIM;
            v16bf b0 = load_b_frag(kb, HDIM);
            v16bf b1 = load_b_frag(kb + 32, HDIM);
            v8f acc = zero8();
            acc = wmma_bf16(a0, b0, acc);
            acc = wmma_bf16(a1, b1, acc);
            float mc = mbuf[(size_t)bh * CTX + sc + col];
            float rc = rbuf[(size_t)bh * CTX + sc + col];
#pragma unroll
            for (int v2 = 0; v2 < 8; ++v2) {
                int t = t0 + v2 + 8 * lh;
                int s = sc + col;
                float p = (t >= s) ? __expf(acc[v2] * SCALE - mc) * rc : 0.0f;
                pb[(v2 + 8 * lh) * 48 + hh * 16 + col] = f2bf_u16(p);
            }
        }
        // intra-wave LDS transpose fence: wait for this wave's DS stores
        // before re-reading the P tile as an A-fragment (cross-lane exchange).
        asm volatile("s_wait_dscnt 0" ::: "memory");
        v16bf ap = load_a_frag(pb, 48);     // P tile 16x32, K = s
#pragma unroll
        for (int nb = 0; nb < 4; ++nb) {
            const unsigned short* vb = vT + ((size_t)bh * HDIM + nb * 16) * CTX + s0;
            v16bf bv = load_b_frag(vb, CTX);
            acco[nb] = wmma_bf16(ap, bv, acco[nb]);
        }
    }

    // write concat-head layout attn[b][t][h*64+d] (bf16)
#pragma unroll
    for (int nb = 0; nb < 4; ++nb)
#pragma unroll
        for (int v2 = 0; v2 < 8; ++v2) {
            int t = t0 + v2 + 8 * lh;
            int d = nb * 16 + col;
            attn[((size_t)(b * CTX + t)) * N_EMB + h * HDIM + d] =
                f2bf_u16(acco[nb][v2]);
        }
}

// ---------------- stage 4: output projection + bias ----------------
__global__ __launch_bounds__(256) void out_proj_kernel(
    const unsigned short* __restrict__ attn,
    const unsigned short* __restrict__ woT,
    const float* __restrict__ bo,
    float* __restrict__ out) {
    __shared__ __align__(16) unsigned short as_[128 * 48];
    __shared__ __align__(16) unsigned short bs_[64 * 48];

    int tid = threadIdx.x;
    int lane = tid & 31, wave = tid >> 5;
    int mbase = blockIdx.x * 128;
    int nbase = blockIdx.y * 64;

    v8f acc[4];
#pragma unroll
    for (int nb = 0; nb < 4; ++nb) acc[nb] = zero8();

    for (int kc = 0; kc < N_EMB; kc += 32) {
        {
            int row = tid >> 1, half = tid & 1;
            const unsigned short* src =
                attn + (size_t)(mbase + row) * N_EMB + kc + half * 16;
            async_copy_32B(src, &as_[row * 48 + half * 16]);
        }
        if (tid < 128) {
            int row = tid >> 1, half = tid & 1;
            const unsigned short* src =
                woT + (size_t)(nbase + row) * N_EMB + kc + half * 16;
            async_copy_32B(src, &bs_[row * 48 + half * 16]);
        }
        async_fence();
        __syncthreads();
        v16bf a = load_a_frag(&as_[wave * 16 * 48], 48);
#pragma unroll
        for (int nb = 0; nb < 4; ++nb) {
            v16bf bf = load_b_frag(&bs_[nb * 16 * 48], 48);
            acc[nb] = wmma_bf16(a, bf, acc[nb]);
        }
        __syncthreads();
    }

    int lh = lane >> 4, col = lane & 15;
#pragma unroll
    for (int nb = 0; nb < 4; ++nb)
#pragma unroll
        for (int v2 = 0; v2 < 8; ++v2) {
            int mm = mbase + wave * 16 + v2 + 8 * lh;
            int nn = nbase + nb * 16 + col;
            out[(size_t)mm * N_EMB + nn] = acc[nb][v2] + bo[nn];
        }
}

// ---------------- workspace layout (bytes) ----------------
#define OFF_XB   ((size_t)0)              // 16 MB  bf16 x [BT][1024]
#define OFF_WQ   ((size_t)16777216)       //  2 MB  bf16 WqT [h][d][c]
#define OFF_WK   ((size_t)18874368)
#define OFF_WV   ((size_t)20971520)
#define OFF_WO   ((size_t)23068672)       //  2 MB  bf16 WoT [n][k]
#define OFF_Q    ((size_t)25165824)       // 16 MB  bf16 q [bh][t][d]
#define OFF_K    ((size_t)41943040)       // 16 MB
#define OFF_VT   ((size_t)58720256)       // 16 MB  bf16 vT [bh][d][t]
#define OFF_M    ((size_t)75497472)       // 512 KB f32 column max
#define OFF_R    ((size_t)76021760)       // 512 KB f32 1/denom
#define OFF_ATT  ((size_t)76546048)       // 16 MB  bf16 concat attn [bt][1024]

extern "C" void kernel_launch(void* const* d_in, const int* in_sizes, int n_in,
                              void* d_out, int out_size, void* d_ws, size_t ws_size,
                              hipStream_t stream) {
    const float* x  = (const float*)d_in[0];
    const float* Wq = (const float*)d_in[1];
    const float* Wk = (const float*)d_in[2];
    const float* Wv = (const float*)d_in[3];
    const float* Wo = (const float*)d_in[4];
    const float* bo = (const float*)d_in[5];
    float* out = (float*)d_out;

    char* ws = (char*)d_ws;
    unsigned short* xb  = (unsigned short*)(ws + OFF_XB);
    unsigned short* wqT = (unsigned short*)(ws + OFF_WQ);
    unsigned short* wkT = (unsigned short*)(ws + OFF_WK);
    unsigned short* wvT = (unsigned short*)(ws + OFF_WV);
    unsigned short* woT = (unsigned short*)(ws + OFF_WO);
    unsigned short* qb  = (unsigned short*)(ws + OFF_Q);
    unsigned short* kb  = (unsigned short*)(ws + OFF_K);
    unsigned short* vTb = (unsigned short*)(ws + OFF_VT);
    float*          mb  = (float*)(ws + OFF_M);
    float*          rb  = (float*)(ws + OFF_R);
    unsigned short* att = (unsigned short*)(ws + OFF_ATT);

    // stage 0: precision conversion / weight transposes
    {
        int n = BT * N_EMB;  // 8388608
        cvt_bf16_kernel<<<(n + 255) / 256, 256, 0, stream>>>(x, xb, n);
        int nw = HEADS * N_EMB * HDIM;  // 1048576
        cvt_transpose_kernel<<<(nw + 255) / 256, 256, 0, stream>>>(Wq, wqT, HEADS, N_EMB, HDIM);
        cvt_transpose_kernel<<<(nw + 255) / 256, 256, 0, stream>>>(Wk, wkT, HEADS, N_EMB, HDIM);
        cvt_transpose_kernel<<<(nw + 255) / 256, 256, 0, stream>>>(Wv, wvT, HEADS, N_EMB, HDIM);
        int no = N_EMB * N_EMB;
        cvt_transpose_kernel<<<(no + 255) / 256, 256, 0, stream>>>(Wo, woT, 1, N_EMB, N_EMB);
    }

    // stage 1: q,k,v projections (v stored transposed)
    qkv_proj_kernel<<<dim3(CTX / 128, BATCH * HEADS), 256, 0, stream>>>(
        xb, wqT, wkT, wvT, qb, kb, vTb);

    // stage 2: column-softmax statistics (softmax over query axis)
    col_stats_kernel<<<(BATCH * HEADS * (CTX / 16)) / 4, 128, 0, stream>>>(qb, kb, mb, rb);

    // stage 3: fused attention output, concat-head bf16
    attn_out_kernel<<<(BATCH * HEADS * (CTX / 16)) / 4, 128, 0, stream>>>(
        qb, kb, vTb, mb, rb, att);

    // stage 4: output projection + bias -> f32
    out_proj_kernel<<<dim3(BT / 128, N_EMB / 64), 256, 0, stream>>>(att, woT, bo, out);
}
